// Floyd_39410619908255
// MI455X (gfx1250) — matrix operand order repaired and checked by reading, not compiled, not run on hardware
//
#include <hip/hip_runtime.h>
#include <stdint.h>

// ---------------------------------------------------------------------------
// Floyd-style permutation-synchronization solver for MI455X (gfx1250).
//
// Exploits the invariant that every X[i,j] is a permutation matrix:
//  * quadratic form vx^T K_ij vx  -> ordered 40x40 gather-sum over K_ij
//  * aff table cached; norm = max of 64 cached values (kills ~9GB of K reads)
//  * _pc -> exact integer mismatch counts
//  * combo einsum (8 x 40x40 matmuls) -> v_wmma_f32_16x16x32_f16, one wave
//    per m', fragments synthesized from permutation tables (exact 0/1 data)
// Sequential 448-step chain -> single persistent 256-thread workgroup.
// ---------------------------------------------------------------------------

#define M_DIM 8
#define N_DIM 40
#define NN    (N_DIM * N_DIM)          // 1600
#define BLK   256
#define KBLK  ((size_t)NN * (size_t)NN) // 2,560,000 floats per (i,j) K block

typedef _Float16 v16h __attribute__((ext_vector_type(16)));
typedef float    v8f  __attribute__((ext_vector_type(8)));

// ---------------------------------------------------------------------------
// Init kernel: aff(i,j) = vx^T K_ij vx for all 64 entries (diag -> 0).
// vx support: p = a*40 + rho[a] ascending in a, where rho[a] = row holding the
// 1 in column a of X[i,j]. Summation order matches the reference's matvec.
// ---------------------------------------------------------------------------
__global__ void __launch_bounds__(BLK)
aff_init_kernel(const float* __restrict__ Kg,
                const float* __restrict__ Xin,
                float* __restrict__ aff_out) {
    __shared__ uint16_t rho[N_DIM];
    __shared__ float    G[N_DIM][N_DIM];
    __shared__ float    RS[N_DIM];

    const int b = blockIdx.x;
    const int i = b >> 3, j = b & 7;
    const int tid = threadIdx.x;

    if (i == j) { if (tid == 0) aff_out[b] = 0.0f; return; }

    const float* Xij = Xin + (size_t)b * NN;
    if (tid < N_DIM) {
        int r = 0;
        for (int rr = 0; rr < N_DIM; ++rr)
            if (Xij[rr * N_DIM + tid] == 1.0f) r = rr;
        rho[tid] = (uint16_t)r;
    }
    __syncthreads();

    const float* Kb = Kg + (size_t)b * KBLK;
    for (int idx = tid; idx < NN; idx += BLK) {
        int a = idx / N_DIM, c = idx - a * N_DIM;
        G[a][c] = Kb[(size_t)(a * N_DIM + rho[a]) * NN + (c * N_DIM + rho[c])];
    }
    __syncthreads();

    if (tid < N_DIM) {                 // inner sum: ascending c (== ascending q)
        float s = 0.0f;
        for (int c = 0; c < N_DIM; ++c) s += G[tid][c];
        RS[tid] = s;
    }
    __syncthreads();
    if (tid == 0) {                    // outer sum: ascending a (== ascending p)
        float s = 0.0f;
        for (int a = 0; a < N_DIM; ++a) s += RS[a];
        aff_out[b] = s;
    }
}

// ---------------------------------------------------------------------------
// WMMA fragment builders (contiguous k order k = 16*(lane/16) + h in A and B, wave32).
// A tile (16x32) of X[i,m']: elem(M,K) = (P_im[M] == K).
// B tile (32x16) of X[m',j]: elem(K,N) = (Pinv_mj[N] == K).
// ---------------------------------------------------------------------------
__device__ __forceinline__ v16h buildA(const uint8_t* Prow, int mt, int kc, int lane) {
    const int row = (lane & 15) + mt * 16;
    const int hi  = lane >> 4;
    const int p   = (row < N_DIM) ? (int)Prow[row] : 255;
    v16h a;
#pragma unroll
    for (int h = 0; h < 16; ++h) {
        const int kk = kc * 32 + (hi ? 16 : 0) + h;
        a[h] = (p == kk) ? (_Float16)1.0f : (_Float16)0.0f;
    }
    return a;
}

__device__ __forceinline__ v16h buildB(const uint8_t* PinvRow, int kc, int nt, int lane) {
    const int col = (lane & 15) + nt * 16;
    const int hi  = lane >> 4;
    const int q   = (col < N_DIM) ? (int)PinvRow[col] : 255;
    v16h bf;
#pragma unroll
    for (int h = 0; h < 16; ++h) {
        const int kk = kc * 32 + (hi ? 16 : 0) + h;
        bf[h] = (q == kk) ? (_Float16)1.0f : (_Float16)0.0f;
    }
    return bf;
}

// ---------------------------------------------------------------------------
// Persistent solver: one workgroup, 8 waves, all state in LDS.
// ---------------------------------------------------------------------------
__global__ void __launch_bounds__(BLK)
floyd_main_kernel(const float* __restrict__ Kg,
                  const float* __restrict__ Xin,
                  const float* __restrict__ affInit,
                  float* __restrict__ Xout) {
    __shared__ uint8_t P[M_DIM * M_DIM][N_DIM];     // row -> col permutation
    __shared__ uint8_t Pinv[M_DIM * M_DIM][N_DIM];  // col -> row permutation
    __shared__ uint8_t CB[M_DIM][NN];               // combo matrices from WMMA
    __shared__ float   G[2][N_DIM][N_DIM];          // gathered K values
    __shared__ float   RS[2][N_DIM];
    __shared__ float   afftab[M_DIM * M_DIM];
    __shared__ uint8_t pic[N_DIM], pinvc[N_DIM];    // pair combo perm / inverse
    __shared__ int     cnt_ori, cnt_new, updFlag;
    __shared__ float   snorm, qf01[2];

    const int tid  = threadIdx.x;
    const int lane = tid & 31;
    const int wave = tid >> 5;                       // 0..7 == m'

    // ---- init LDS state from inputs ----
    for (int t = tid; t < M_DIM * M_DIM; t += BLK) afftab[t] = affInit[t];
    for (int t = tid; t < M_DIM * M_DIM * N_DIM; t += BLK) {
        int pr = t / N_DIM, r = t - pr * N_DIM;
        const float* Xr = Xin + (size_t)pr * NN + (size_t)r * N_DIM;
        int c = 0;
        for (int cc = 0; cc < N_DIM; ++cc)
            if (Xr[cc] == 1.0f) c = cc;
        P[pr][r] = (uint8_t)c;
    }
    __syncthreads();
    for (int t = tid; t < M_DIM * M_DIM * N_DIM; t += BLK) {
        int pr = t / N_DIM, r = t - pr * N_DIM;
        Pinv[pr][P[pr][r]] = (uint8_t)r;
    }
    __syncthreads();

    for (int ph = 0; ph < 2; ++ph) {
        const float cst = ph ? 0.3f : 0.0f;
        for (int k = 0; k < M_DIM; ++k) {
            if (tid == 0) {                          // norm = max of cached affs
                float mx = afftab[0];
                for (int t = 1; t < M_DIM * M_DIM; ++t) mx = fmaxf(mx, afftab[t]);
                snorm = mx;
            }
            __syncthreads();

            for (int i = 0; i < M_DIM - 1; ++i) {
                for (int j = i + 1; j < M_DIM; ++j) {
                    // (1) combo perm: pi_c = P_kj o P_ik ; and its inverse
                    if (tid < N_DIM) {
                        pic[tid]   = P[k * M_DIM + j][P[i * M_DIM + k][tid]];
                        pinvc[tid] = Pinv[i * M_DIM + k][Pinv[k * M_DIM + j][tid]];
                    }
                    if (tid == 0) { cnt_ori = 0; cnt_new = 0; }
                    __syncthreads();

                    // (2) issue both speculative gathers into registers:
                    //     which=0: K[i,j] with rho = pi_c^-1  (X[i,j]=combo)
                    //     which=1: K[j,i] with rho = pi_c     (X[j,i]=combo^T)
                    float gv[13];
                    int gn = 0;
#pragma unroll
                    for (int u = 0; u < 13; ++u) {
                        const int idx = tid + u * BLK;
                        if (idx < 2 * NN) {
                            const int which = idx / NN;
                            const int rem = idx - which * NN;
                            const int a = rem / N_DIM, c = rem - a * N_DIM;
                            const uint8_t* rho = which ? pic : pinvc;
                            const int bi = which ? (j * M_DIM + i) : (i * M_DIM + j);
                            const float* Kb = Kg + (size_t)bi * KBLK;
                            gv[u] = Kb[(size_t)(a * N_DIM + rho[a]) * NN +
                                       (c * N_DIM + rho[c])];
                            gn = u + 1;
                        }
                    }

                    // (3) WMMA: wave computes combo_{m'=wave} = X[i,m'] @ X[m',j]
                    {
                        const uint8_t* Pim = P[i * M_DIM + wave];
                        const uint8_t* Qmj = Pinv[wave * M_DIM + j];
#pragma unroll
                        for (int mt = 0; mt < 3; ++mt) {
#pragma unroll
                            for (int nt = 0; nt < 3; ++nt) {
                                v8f acc = {0.0f, 0.0f, 0.0f, 0.0f,
                                           0.0f, 0.0f, 0.0f, 0.0f};
#pragma unroll
                                for (int kc = 0; kc < 2; ++kc) {
                                    v16h av = buildA(Pim, mt, kc, lane);
                                    v16h bv = buildB(Qmj, kc, nt, lane);
                                    acc = __builtin_amdgcn_wmma_f32_16x16x32_f16(
                                        false, av, false, bv,
                                        (short)0, acc, false, false);
                                }
                                const int colN  = nt * 16 + (lane & 15);
                                const int rbase = mt * 16 + ((lane >> 4) ? 8 : 0);
#pragma unroll
                                for (int e = 0; e < 8; ++e) {
                                    const int rowM = rbase + e;
                                    if (rowM < N_DIM && colN < N_DIM)
                                        CB[wave][rowM * N_DIM + colN] =
                                            (uint8_t)(acc[e] != 0.0f ? 1 : 0);
                                }
                            }
                        }
                    }
                    __syncthreads();

                    // (4) park gathers in LDS + exact mismatch counts from WMMA
                    for (int u = 0; u < gn; ++u) {
                        const int idx = tid + u * BLK;
                        const int which = idx / NN;
                        const int rem = idx - which * NN;
                        G[which][rem / N_DIM][rem % N_DIM] = gv[u];
                    }
                    {
                        int co = 0, cn = 0;
                        const uint8_t* Pij = P[i * M_DIM + j];
                        for (int t = tid; t < M_DIM * NN; t += BLK) {
                            const int mm = t / NN, rc = t - mm * NN;
                            const int r = rc / N_DIM, c = rc - r * N_DIM;
                            const uint8_t v   = CB[mm][rc];
                            const uint8_t ori = (Pij[r] == c) ? 1 : 0;
                            const uint8_t nw  = CB[k][rc];   // pair combo == m'=k
                            co += (v != ori);
                            cn += (v != nw);
                        }
                        if (co) atomicAdd(&cnt_ori, co);
                        if (cn) atomicAdd(&cnt_new, cn);
                    }
                    __syncthreads();

                    // (5) ordered reductions (match reference matvec nesting)
                    if (tid < 2 * N_DIM) {
                        const int w = tid / N_DIM, a = tid - w * N_DIM;
                        float s = 0.0f;
                        for (int c = 0; c < N_DIM; ++c) s += G[w][a][c];
                        RS[w][a] = s;
                    }
                    __syncthreads();
                    if (tid < 2) {
                        float s = 0.0f;
                        for (int a = 0; a < N_DIM; ++a) s += RS[tid][a];
                        qf01[tid] = s;
                    }
                    __syncthreads();

                    // (6) decision
                    if (tid == 0) {
                        const float norm = snorm;
                        const float aff_o = afftab[i * M_DIM + j] / norm;
                        const float aff_n = qf01[0] / norm;
                        const float pc_o = 1.0f - (float)cnt_ori / 640.0f;
                        const float pc_n = 1.0f - (float)cnt_new / 640.0f;
                        const float s_o = aff_o * (1.0f - cst) + pc_o * cst;
                        const float s_n = aff_n * (1.0f - cst) + pc_n * cst;
                        const int upd = (s_n > s_o) ? 1 : 0;
                        updFlag = upd;
                        if (upd) {
                            afftab[i * M_DIM + j] = qf01[0];
                            afftab[j * M_DIM + i] = qf01[1];
                        }
                    }
                    __syncthreads();

                    // (7) apply update: X[i,j]=combo, X[j,i]=combo^T (as perms)
                    if (updFlag && tid < N_DIM) {
                        P[i * M_DIM + j][tid]    = pic[tid];
                        Pinv[i * M_DIM + j][tid] = pinvc[tid];
                        P[j * M_DIM + i][tid]    = pinvc[tid];
                        Pinv[j * M_DIM + i][tid] = pic[tid];
                    }
                    __syncthreads();
                }
            }
        }
    }

    // ---- materialize fp32 output ----
    for (int t = tid; t < M_DIM * M_DIM * NN; t += BLK) {
        const int pr = t / NN, rc = t - pr * NN;
        const int r = rc / N_DIM, c = rc - r * N_DIM;
        Xout[t] = (P[pr][r] == c) ? 1.0f : 0.0f;
    }
}

// ---------------------------------------------------------------------------
extern "C" void kernel_launch(void* const* d_in, const int* in_sizes, int n_in,
                              void* d_out, int out_size, void* d_ws, size_t ws_size,
                              hipStream_t stream) {
    (void)in_sizes; (void)n_in; (void)out_size;
    const float* Kg  = (const float*)d_in[0];   // (8,8,1600,1600) f32
    const float* Xin = (const float*)d_in[1];   // (8,8,40,40) f32
    float* out = (float*)d_out;

    // aff table scratch (64 floats): use workspace, fall back to d_out's head
    // (main kernel reads it before writing any output).
    float* aff = (ws_size >= 64 * sizeof(float)) ? (float*)d_ws : out;

    aff_init_kernel<<<64, BLK, 0, stream>>>(Kg, Xin, aff);
    floyd_main_kernel<<<1, BLK, 0, stream>>>(Kg, Xin, aff, out);
}
